// MultiheadLMAttentionWithCache_76776835383894
// MI455X (gfx1250) — matrix operand order repaired and checked by reading, not compiled, not run on hardware
//
#include <hip/hip_runtime.h>

// ---------------- problem constants (fixed by reference) ----------------
#define T_LEN   1024
#define BSZ     4
#define EMB     2048
#define NH      16
#define HD      128
#define SMAX    2048
#define CLEN    1024
#define BH      64          // BSZ*NH
#define MROWS   4096        // T*B
#define N_IN    6144        // 3*E
#define SCALING 0.08838834764831845f  // 128^-0.5
#define NEGBIG  (-3.0e38f)

typedef __attribute__((ext_vector_type(16))) _Float16 v16h;
typedef __attribute__((ext_vector_type(8)))  float    v8f;

// gfx1250 async global->LDS path (guarded so either toolchain compiles)
#if __has_builtin(__builtin_amdgcn_global_load_async_to_lds_b128) && \
    __has_builtin(__builtin_amdgcn_s_wait_asynccnt)
#define USE_ASYNC_LDS 1
typedef int v4i __attribute__((ext_vector_type(4)));
typedef __attribute__((address_space(1))) v4i g_v4i;   // global int4
typedef __attribute__((address_space(3))) v4i l_v4i;   // LDS int4
#else
#define USE_ASYNC_LDS 0
#endif

// ---------------- WMMA wrapper (codegen-confirmed builtin) ----------------
static __device__ __forceinline__ v8f wmma16(v16h a, v16h b, v8f c) {
  // D = A(16x32 f16) * B(32x16 f16) + C(16x16 f32)
  return __builtin_amdgcn_wmma_f32_16x16x32_f16(
      /*neg_a=*/false, a, /*neg_b=*/false, b,
      /*c_mod=*/(short)0, c, /*reuse_a=*/false, /*reuse_b=*/false);
}

// ---- fragment loaders (wave32; contiguous k = 16*half + s, same order in A and B) ----
// A 16x32 f16: lane m16 holds row m16; 16 contiguous K values starting at half*16
static __device__ __forceinline__ v16h load_a_f32cvt(const float* __restrict__ row,
                                                     int k, int half) {
  union { v16h v; _Float16 h[16]; } f;
  const float* p0 = row + k + half * 16;
  const float* p1 = row + k + half * 16 + 8;
#pragma unroll
  for (int i = 0; i < 8; ++i) { f.h[i] = (_Float16)p0[i]; f.h[8 + i] = (_Float16)p1[i]; }
  return f.v;
}
static __device__ __forceinline__ v16h load_a_f16(const _Float16* __restrict__ row,
                                                  int k, int half) {
  union { v16h v; _Float16 h[16]; } f;
  const _Float16* p0 = row + k + half * 16;
  const _Float16* p1 = row + k + half * 16 + 8;
#pragma unroll
  for (int i = 0; i < 8; ++i) { f.h[i] = p0[i]; f.h[8 + i] = p1[i]; }
  return f.v;
}
// B 32x16 f16: lane m16 = column n; 16 contiguous K values starting at half*16
static __device__ __forceinline__ v16h load_b_f32cvt(const float* __restrict__ p) {
  union { v16h v; _Float16 h[16]; } f;
#pragma unroll
  for (int i = 0; i < 16; ++i) f.h[i] = (_Float16)p[i];
  return f.v;
}

// =====================================================================
// K0: copy old cache rows [0, CLEN) of cache_k / cache_v to outputs
// =====================================================================
__global__ __launch_bounds__(256) void k_copy_old(const float4* __restrict__ ki,
                                                  const float4* __restrict__ vi,
                                                  float4* __restrict__ ko,
                                                  float4* __restrict__ vo) {
  const size_t per = (size_t)BH * CLEN * (HD / 4);  // float4 per array
  size_t idx = (size_t)blockIdx.x * blockDim.x + threadIdx.x;
  if (idx >= 2 * per) return;
  bool isV = idx >= per;
  size_t i = isV ? idx - per : idx;
  size_t bh = i / (CLEN * (HD / 4));
  size_t rem = i % (CLEN * (HD / 4));
  size_t off = bh * (size_t)(SMAX * (HD / 4)) + rem;
  if (isV) vo[off] = vi[off];
  else     ko[off] = ki[off];
}

// =====================================================================
// K1: proj GEMM  Y[4096,6144] = x @ w_in^T + b_in, scatter q/k/v
//     wave tile 64x64, block = 4 waves -> 64x256, grid (64, 24)
// =====================================================================
__global__ __launch_bounds__(128) void k_gemm_qkv(const float* __restrict__ x,
                                                  const float* __restrict__ w_in,
                                                  const float* __restrict__ b_in,
                                                  _Float16* __restrict__ qF,
                                                  float* __restrict__ ck,
                                                  float* __restrict__ cv) {
  const int wave = threadIdx.x >> 5, lane = threadIdx.x & 31;
  const int m16 = lane & 15, half = lane >> 4;
  const int m0 = blockIdx.x * 64;
  const int n0 = blockIdx.y * 256 + wave * 64;

  v8f vz = {};
  v8f acc[4][4];
#pragma unroll
  for (int i = 0; i < 4; ++i)
#pragma unroll
    for (int j = 0; j < 4; ++j) acc[i][j] = vz;

  for (int k = 0; k < EMB; k += 32) {
    v16h a[4];
#pragma unroll
    for (int i = 0; i < 4; ++i) {
      const float* row = x + (size_t)(m0 + i * 16 + m16) * EMB;
      __builtin_prefetch(row + k + 32, 0, 3);  // unconditional: branch-free global_prefetch_b8
      a[i] = load_a_f32cvt(row, k, half);
    }
#pragma unroll
    for (int j = 0; j < 4; ++j) {
      const float* brow = w_in + (size_t)(n0 + j * 16 + m16) * EMB + k + half * 16;
      __builtin_prefetch(brow + 32, 0, 3);
      v16h bf = load_b_f32cvt(brow);
#pragma unroll
      for (int i = 0; i < 4; ++i) acc[i][j] = wmma16(a[i], bf, acc[i][j]);
    }
  }

  // epilogue: col n -> (head h, section, dim d); row m -> (t, b)
#pragma unroll
  for (int j = 0; j < 4; ++j) {
    const int n = n0 + j * 16 + m16;         // 16-wide tile stays inside one section
    const int h = n / 384;
    const int sct = (n % 384) / HD;
    const int d = n % HD;
    const float bias = b_in[n];
#pragma unroll
    for (int i = 0; i < 4; ++i) {
#pragma unroll
      for (int vv = 0; vv < 8; ++vv) {
        const int m = m0 + i * 16 + vv + 8 * half;
        const int t = m >> 2, bb = m & 3;
        const int bh = bb * NH + h;
        const float val = acc[i][j][vv] + bias;
        if (sct == 0)
          qF[((size_t)bh * T_LEN + t) * HD + d] = (_Float16)(val * SCALING);
        else if (sct == 1)
          ck[((size_t)bh * SMAX + CLEN + t) * HD + d] = val;
        else
          cv[((size_t)bh * SMAX + CLEN + t) * HD + d] = val;
      }
    }
  }
}

// =====================================================================
// K2: transpose updated V cache to f16:  vT[bh][d][p] = (f16) cv[bh][p][d]
//     32x32 LDS tile transpose, block (32,8), grid (64, 4, 64)
// =====================================================================
__global__ __launch_bounds__(256) void k_transpose_v(const float* __restrict__ cv,
                                                     _Float16* __restrict__ vT) {
  __shared__ _Float16 tile[32][33];
  const int bh = blockIdx.z;
  const int p0 = blockIdx.x * 32, d0 = blockIdx.y * 32;
  for (int yy = threadIdx.y; yy < 32; yy += 8)
    tile[yy][threadIdx.x] =
        (_Float16)cv[((size_t)bh * SMAX + p0 + yy) * HD + d0 + threadIdx.x];
  __syncthreads();
  for (int yy = threadIdx.y; yy < 32; yy += 8)
    vT[((size_t)bh * HD + d0 + yy) * SMAX + p0 + threadIdx.x] = tile[threadIdx.x][yy];
}

// =====================================================================
// K3: flash attention. Block = 4 waves sharing ONE head, 4 consecutive
//     16-row q tiles -> K tile staged once per block into LDS via the
//     gfx1250 async global->LDS pipeline. grid = 64 heads * 16 qgroups.
// =====================================================================
__global__ __launch_bounds__(128) void k_attention(const _Float16* __restrict__ qF,
                                                   const float* __restrict__ ck,
                                                   const _Float16* __restrict__ vT,
                                                   _Float16* __restrict__ attnF) {
  __shared__ float    ktile[32 * HD];    // 16KB: staged K tile (32 kv rows x 128 d)
  __shared__ _Float16 pbuf[4][16 * 32];  // per-wave P staging (16x32 f16)
  const int wave = threadIdx.x >> 5, lane = threadIdx.x & 31;
  const int m16 = lane & 15, half = lane >> 4;
  const int bh = blockIdx.x >> 4;        // head (0..63) shared by whole block
  const int qg = blockIdx.x & 15;        // q group (64 rows)
  const int qt = qg * 4 + wave;          // this wave's 16-row q tile
  const int b_ = bh >> 4, h_ = bh & 15;

  // Q A-fragments (16 rows x 128, K-split into 4 x 32), already scaled
  const _Float16* qrow = qF + ((size_t)bh * T_LEN + qt * 16 + m16) * HD;
  v16h aq[4];
#pragma unroll
  for (int kk = 0; kk < 4; ++kk) aq[kk] = load_a_f16(qrow, kk * 32, half);

  v8f vz = {};
  v8f o[8];
#pragma unroll
  for (int dt = 0; dt < 8; ++dt) o[dt] = vz;
  float rmax[8], rsum[8];
#pragma unroll
  for (int vv = 0; vv < 8; ++vv) { rmax[vv] = NEGBIG; rsum[vv] = 0.0f; }

  const float* kbase = ck + (size_t)bh * SMAX * HD;
  const _Float16* vbase = vT + (size_t)bh * HD * SMAX;
  _Float16* pb = &pbuf[wave][0];
  // block-uniform trip count = widest wave in the block (masking handles the rest)
  const int kv_end = CLEN + (qg * 4 + 3) * 16 + 16;

  for (int p0 = 0; p0 < kv_end; p0 += 32) {
    // ---- cooperative stage of K tile (32 rows x 128 f32) into LDS ----
#pragma unroll
    for (int c = 0; c < 8; ++c) {
      const int chunk = c * 128 + threadIdx.x;   // 1024 x 16B chunks
      const int row = chunk >> 5;                // 32 chunks per 128-float row
      const int col = (chunk & 31) * 4;
      const float* g = kbase + (size_t)(p0 + row) * HD + col;
#if USE_ASYNC_LDS
      __builtin_amdgcn_global_load_async_to_lds_b128(
          (g_v4i*)g, (l_v4i*)&ktile[row * HD + col], 0, 0);
#else
      *(float4*)&ktile[row * HD + col] = *(const float4*)g;
#endif
    }
#if USE_ASYNC_LDS
    __builtin_amdgcn_s_wait_asynccnt(0);
#endif
    __syncthreads();

    // ---- S = Q K^T for this wave's 16x32 tile (two 16x16 halves) ----
    v8f s0 = vz, s1 = vz;
#pragma unroll
    for (int kk = 0; kk < 4; ++kk) {
      const int doff = kk * 32 + half * 16;
      v16h b0 = load_b_f32cvt(ktile + (size_t)m16 * HD + doff);
      s0 = wmma16(aq[kk], b0, s0);
      v16h b1 = load_b_f32cvt(ktile + (size_t)(16 + m16) * HD + doff);
      s1 = wmma16(aq[kk], b1, s1);
    }

    // ---- causal mask + online softmax (row = vv+8*half, col = m16) ----
    float pr0[8], pr1[8];
#pragma unroll
    for (int vv = 0; vv < 8; ++vv) {
      const int t = qt * 16 + vv + 8 * half;
      const int limit = CLEN + t;
      float a0 = (p0 + m16      <= limit) ? s0[vv] : NEGBIG;
      float a1 = (p0 + 16 + m16 <= limit) ? s1[vv] : NEGBIG;
      float mcur = fmaxf(a0, a1);
#pragma unroll
      for (int off = 8; off; off >>= 1) mcur = fmaxf(mcur, __shfl_xor(mcur, off, 16));
      const float nm = fmaxf(rmax[vv], mcur);
      const float sc = __expf(rmax[vv] - nm);
      rmax[vv] = nm;
      const float e0 = __expf(a0 - nm);
      const float e1 = __expf(a1 - nm);
      float rs = e0 + e1;
#pragma unroll
      for (int off = 8; off; off >>= 1) rs += __shfl_xor(rs, off, 16);
      rsum[vv] = rsum[vv] * sc + rs;
      pr0[vv] = e0; pr1[vv] = e1;
#pragma unroll
      for (int dt = 0; dt < 8; ++dt) o[dt][vv] *= sc;  // rescale running output
    }

    // ---- relayout P (C-layout -> A-layout) through LDS (per-wave region) ----
#pragma unroll
    for (int vv = 0; vv < 8; ++vv) {
      const int m = vv + 8 * half;
      pb[m * 32 + m16]      = (_Float16)pr0[vv];
      pb[m * 32 + 16 + m16] = (_Float16)pr1[vv];
    }
    __builtin_amdgcn_wave_barrier();  // same-wave DS ops are in-order; block reordering
    v16h ap = load_a_f16(pb + m16 * 32, 0, half);
    __builtin_amdgcn_wave_barrier();

    // ---- O += P @ V  (B-frag: 16 contiguous f16 along p in vT) ----
#pragma unroll
    for (int dt = 0; dt < 8; ++dt) {
      const _Float16* vp = vbase + (size_t)(dt * 16 + m16) * SMAX + p0 + half * 16;
      v16h bv = *(const v16h*)vp;  // 32B-aligned by construction
      o[dt] = wmma16(ap, bv, o[dt]);
    }
    __syncthreads();  // ktile reused next iteration
  }

  // ---- epilogue: attnF[t*B+b][h*128+d] = O / rowsum ----
#pragma unroll
  for (int dt = 0; dt < 8; ++dt) {
#pragma unroll
    for (int vv = 0; vv < 8; ++vv) {
      const int m = vv + 8 * half;
      const int t = qt * 16 + m;
      const int d = dt * 16 + m16;
      const float val = o[dt][vv] / rsum[vv];
      attnF[(size_t)(t * BSZ + b_) * EMB + h_ * HD + d] = (_Float16)val;
    }
  }
}

// =====================================================================
// K4: out GEMM  out[4096,2048] = attnF @ w_out^T + b_out
//     wave tile 64x64, block = 4 waves, grid (64, 8)
// =====================================================================
__global__ __launch_bounds__(128) void k_gemm_out(const _Float16* __restrict__ attnF,
                                                  const float* __restrict__ w_out,
                                                  const float* __restrict__ b_out,
                                                  float* __restrict__ out) {
  const int wave = threadIdx.x >> 5, lane = threadIdx.x & 31;
  const int m16 = lane & 15, half = lane >> 4;
  const int m0 = blockIdx.x * 64;
  const int n0 = blockIdx.y * 256 + wave * 64;

  v8f vz = {};
  v8f acc[4][4];
#pragma unroll
  for (int i = 0; i < 4; ++i)
#pragma unroll
    for (int j = 0; j < 4; ++j) acc[i][j] = vz;

  for (int k = 0; k < EMB; k += 32) {
    v16h a[4];
#pragma unroll
    for (int i = 0; i < 4; ++i) {
      const _Float16* row = attnF + (size_t)(m0 + i * 16 + m16) * EMB;
      a[i] = load_a_f16(row, k, half);
    }
#pragma unroll
    for (int j = 0; j < 4; ++j) {
      const float* brow = w_out + (size_t)(n0 + j * 16 + m16) * EMB + k + half * 16;
      __builtin_prefetch(brow + 32, 0, 3);
      v16h bf = load_b_f32cvt(brow);
#pragma unroll
      for (int i = 0; i < 4; ++i) acc[i][j] = wmma16(a[i], bf, acc[i][j]);
    }
  }

#pragma unroll
  for (int j = 0; j < 4; ++j) {
    const int n = n0 + j * 16 + m16;
    const float bias = b_out[n];
#pragma unroll
    for (int i = 0; i < 4; ++i) {
#pragma unroll
      for (int vv = 0; vv < 8; ++vv) {
        const int m = m0 + i * 16 + vv + 8 * half;
        out[(size_t)m * EMB + n] = acc[i][j][vv] + bias;
      }
    }
  }
}

// =====================================================================
extern "C" void kernel_launch(void* const* d_in, const int* in_sizes, int n_in,
                              void* d_out, int out_size, void* d_ws, size_t ws_size,
                              hipStream_t stream) {
  const float* x     = (const float*)d_in[0];
  const float* ck_in = (const float*)d_in[1];
  const float* cv_in = (const float*)d_in[2];
  const float* w_in  = (const float*)d_in[3];
  const float* b_in  = (const float*)d_in[4];
  const float* w_out = (const float*)d_in[5];
  const float* b_out = (const float*)d_in[6];
  (void)in_sizes; (void)n_in; (void)out_size; (void)ws_size;

  float* out    = (float*)d_out;                       // (T,B,E)
  float* ck_out = out + (size_t)MROWS * EMB;           // (BH,SMAX,HD)
  float* cv_out = ck_out + (size_t)BH * SMAX * HD;

  // workspace: qF (f16, 16MB) | vT (f16, 32MB) | attnF (f16, 16MB) = 64MB
  _Float16* qF    = (_Float16*)d_ws;
  _Float16* vT    = qF + (size_t)BH * T_LEN * HD;
  _Float16* attnF = vT + (size_t)BH * HD * SMAX;

  // K0: carry over old cache rows
  {
    const size_t n4 = 2ull * BH * CLEN * (HD / 4);
    k_copy_old<<<dim3((unsigned)((n4 + 255) / 256)), dim3(256), 0, stream>>>(
        (const float4*)ck_in, (const float4*)cv_in, (float4*)ck_out, (float4*)cv_out);
  }
  // K1: qkv projection GEMM + scatter (writes qF, new cache rows)
  k_gemm_qkv<<<dim3(MROWS / 64, N_IN / 256), dim3(128), 0, stream>>>(
      x, w_in, b_in, qF, ck_out, cv_out);
  // K2: transpose V cache to f16 (p-major per d) for WMMA B operands
  k_transpose_v<<<dim3(SMAX / 32, HD / 32, BH), dim3(32, 8), 0, stream>>>(cv_out, vT);
  // K3: flash attention (block = one head, 64 q rows, async K staging)
  k_attention<<<dim3(BH * (T_LEN / 64)), dim3(128), 0, stream>>>(qF, ck_out, vT, attnF);
  // K4: output projection GEMM
  k_gemm_out<<<dim3(MROWS / 64, EMB / 256), dim3(128), 0, stream>>>(attnF, w_out, b_out, out);
}
